// LSTMCRF_61804579389736
// MI455X (gfx1250) — matrix operand order repaired and checked by reading, compiled, not run on hardware
//
#include <hip/hip_runtime.h>
#include <hip/hip_bf16.h>

// ---------------------------------------------------------------------------
// BiLSTM + CRF for MI455X (gfx1250, wave32, WMMA).
// All GEMMs use v_wmma_f32_16x16x32_bf16 (f32 accumulate).
//   k_prep     : convert/pad weights to bf16 WMMA-friendly layouts
//   k_embed    : gather embeddings -> bf16, K padded 300->320
//   k_gemm_pre : preT[d][t][g][b] = (e @ W_ih^T + bias)  (WMMA, async-LDS A)
//   k_lstm     : sequential recurrence; W_hh persistent in VGPRs,
//                h double-buffered in LDS, c in LDS, v_tanh_f32 gates
//   k_decoder  : decoded = [hf|hb] @ dec_W^T + dec_b (WMMA)
//   k_crf/k_viterbi/k_reduce : CRF scans -> d_out
// ---------------------------------------------------------------------------

typedef __bf16 bf16_t;
typedef __attribute__((ext_vector_type(16))) bf16_t v16bf;
typedef __attribute__((ext_vector_type(8)))  float  v8f;

#define NV 50000
#define NE 300
#define NEP 320            // E padded to multiple of 32
#define NH 256
#define NG 1024            // 4*H
#define NL 20
#define NB 128
#define NT 512
#define NTOK (NB*NT)       // 65536

// ---- WMMA fragment helpers (bf16 A: 16x32, B: 32x16, C/D: f32 16x16) ------

__device__ __forceinline__ v8f wmma_bf16(v16bf a, v16bf b, v8f c) {
  return __builtin_amdgcn_wmma_f32_16x16x32_bf16(false, a, false, b,
                                                 (short)0, c, false, false);
}

// A fragment: lane(0-15)=M row, lanes 16-31 carry K+8.
__device__ __forceinline__ v16bf load_frag_a(const bf16_t* buf, long ld,
                                             long row0, int k0) {
  int lane = threadIdx.x & 31;
  int half = lane >> 4, mr = lane & 15;
  const bf16_t* p = buf + (row0 + mr) * ld + k0 + half * 8;
  v16bf f;
#pragma unroll
  for (int v = 0; v < 8; ++v) {
    int kk = ((v & 3) * 2) + ((v >> 2) * 16);
    f[2 * v]     = p[kk];
    f[2 * v + 1] = p[kk + 1];
  }
  return f;
}

// B fragment from row-major weight W[g][k]; lane(0-15)=N col g0+n.
__device__ __forceinline__ v16bf load_frag_b(const bf16_t* w, long ld,
                                             int g0, int k0) {
  int lane = threadIdx.x & 31;
  int half = lane >> 4, n = lane & 15;
  const bf16_t* p = w + (long)(g0 + n) * ld + k0 + half * 8;
  v16bf f;
#pragma unroll
  for (int v = 0; v < 8; ++v) {
    int kk = ((v & 3) * 2) + ((v >> 2) * 16);
    f[2 * v]     = p[kk];
    f[2 * v + 1] = p[kk + 1];
  }
  return f;
}

// CDNA5 single-instruction transcendental tanh (TRANS32 op).
// Trailing v_nop covers the trans-result-use hazard (ISA 7.4) since the
// compiler cannot schedule around opaque asm.
__device__ __forceinline__ float fast_tanh(float x) {
  float r;
  asm("v_tanh_f32 %0, %1\n\tv_nop" : "=v"(r) : "v"(x));
  return r;
}
__device__ __forceinline__ float fast_sig(float x) {
  return __builtin_fmaf(0.5f, fast_tanh(0.5f * x), 0.5f);
}
__device__ __forceinline__ float sigf(float x) {
  return 1.0f / (1.0f + __expf(-x));
}

// gfx1250 async global->LDS copy, 16B per lane (ASYNCcnt-tracked)
__device__ __forceinline__ void async_g2l_b128(unsigned lds_off, const void* g) {
  asm volatile("global_load_async_to_lds_b128 %0, %1, off"
               :: "v"(lds_off), "v"(g) : "memory");
}
__device__ __forceinline__ void wait_async0() {
  asm volatile("s_wait_asynccnt 0" ::: "memory");
}

// ---- K0: weight conversion -------------------------------------------------
__global__ void k_prep(const float* Wih_f, const float* Whh_f,
                       const float* Wih_b, const float* Whh_b,
                       const float* decW,
                       bf16_t* WihP, bf16_t* WhhP, bf16_t* decWP) {
  int i = blockIdx.x * blockDim.x + threadIdx.x;
  if (i < 2 * NG * NEP) {               // [2][1024][320], zero-padded K
    int d = i / (NG * NEP); int rem = i % (NG * NEP);
    int g = rem / NEP, k = rem % NEP;
    const float* W = d ? Wih_b : Wih_f;
    WihP[i] = (k < NE) ? (bf16_t)W[(long)g * NE + k] : (bf16_t)0.0f;
  }
  if (i < 2 * NG * NH) {                // [2][1024][256]
    int d = i / (NG * NH); int rem = i % (NG * NH);
    const float* W = d ? Whh_b : Whh_f;
    WhhP[i] = (bf16_t)W[rem];
  }
  if (i < 32 * 2 * NH) {                // [32][512], rows >= 20 zero
    int l = i / (2 * NH), k = i % (2 * NH);
    decWP[i] = (l < NL) ? (bf16_t)decW[(long)l * 2 * NH + k] : (bf16_t)0.0f;
  }
}

// ---- K1: embedding gather, f32 -> bf16, K padded ---------------------------
__global__ void k_embed(const int* x, const float* emb, bf16_t* e) {
  long r = blockIdx.x;                  // token row r = b*T + t
  int k = threadIdx.x;                  // 0..319
  int tok = x[r];
  e[r * NEP + k] = (k < NE) ? (bf16_t)emb[(long)tok * NE + k] : (bf16_t)0.0f;
}

// ---- K2: input projection GEMM ---------------------------------------------
// preT[d][t][g][b] = sum_k e[b*T+t][k] * W_ih[g][k] + bias[g]
// grid (512, 16, 2), block 256 (8 waves). Block tile 128M x 64N, K=320.
// A block staged to LDS with async global->LDS copies. Note: a 128-row
// M-tile spans a single batch b (128 | 512), so b is block-constant.
__global__ void k_gemm_pre(const bf16_t* __restrict__ e,
                           const bf16_t* __restrict__ WihP,
                           const float* __restrict__ b_f,
                           const float* __restrict__ b_b,
                           float* __restrict__ preT) {
  extern __shared__ char smem_raw[];
  bf16_t* sA = (bf16_t*)smem_raw;       // [128][320] bf16 = 80 KB
  int d = blockIdx.z;
  long r0 = (long)blockIdx.x * 128;
  int g_base = blockIdx.y * 64;

  // async stage the A block (rows contiguous: ld == 320)
  const char* src = (const char*)(e + r0 * NEP);
  for (int i = threadIdx.x; i < 128 * NEP * 2 / 16; i += blockDim.x)
    async_g2l_b128((unsigned)(i * 16), src + (size_t)i * 16);
  wait_async0();
  __syncthreads();

  const bf16_t* W = WihP + (long)d * NG * NEP;
  const float* bias = d ? b_b : b_f;
  float* out = preT + (long)d * NT * NG * NB;

  int w = threadIdx.x >> 5;
  int m0 = w * 16;
  int lane = threadIdx.x & 31;
  int n = lane & 15, mh = lane >> 4;
  int bconst = (int)(r0 >> 9);          // batch index of this M block
  int tbase = (int)(r0 & 511);          // time offset of row 0 in block

  for (int nt = 0; nt < 4; ++nt) {
    int g0 = g_base + nt * 16;
    v8f acc = {};
#pragma unroll
    for (int kt = 0; kt < NEP / 32; ++kt) {
      v16bf a  = load_frag_a(sA, NEP, m0, kt * 32);
      v16bf bm = load_frag_b(W, NEP, g0, kt * 32);
      acc = wmma_bf16(a, bm, acc);
    }
    int g = g0 + n;
    float bv = bias[g];
#pragma unroll
    for (int rr = 0; rr < 8; ++rr) {
      int tt = tbase + m0 + mh * 8 + rr;
      out[((long)tt * NG + g) * NB + bconst] = acc[rr] + bv;
    }
  }
}

// ---- K3: sequential BiLSTM recurrence --------------------------------------
// grid 2 (direction), block 512 (16 waves).
// Wave w owns j-tile w (16 gate-aligned columns) for ALL batch rows:
//   - its 32 W_hh fragments (4 gates x 8 k-steps) stay PERSISTENT in VGPRs
//     for the whole 512-step loop (W_hh read from global exactly once),
//   - i/f/g/o of a cell land in the same lane/slot across the 4 accumulators,
//   - preT layout makes the bias/pre additions coalesced float4 loads,
//   - h double-buffered in LDS (128 KB), c in LDS (128 KB),
//   - gates use single-instruction v_tanh_f32.
__global__ void __launch_bounds__(512, 1)
k_lstm(const float* __restrict__ preT, const bf16_t* __restrict__ WhhP,
       bf16_t* __restrict__ hf, bf16_t* __restrict__ hb) {
  extern __shared__ char smem_raw[];
  bf16_t* hbuf = (bf16_t*)smem_raw;                    // [2][128][256]
  float*  c_sh = (float*)(smem_raw + 2 * NB * NH * 2); // [128][256]
  int d = blockIdx.x;
  const bf16_t* W = WhhP + (long)d * NG * NH;
  const float* pr = preT + (long)d * NT * NG * NB;
  bf16_t* hout = d ? hb : hf;

  for (int i = threadIdx.x; i < 2 * NB * NH; i += blockDim.x)
    hbuf[i] = (bf16_t)0.0f;
  for (int i = threadIdx.x; i < NB * NH; i += blockDim.x)
    c_sh[i] = 0.0f;
  __syncthreads();

  int w = threadIdx.x >> 5;             // 0..15 = j-tile
  int lane = threadIdx.x & 31;
  int j0 = w * 16;
  int n = lane & 15, mh = lane >> 4;
  int j = j0 + n;

  // persistent W_hh fragments for this wave's j-tile
  v16bf bfrag[4][8];
#pragma unroll
  for (int gi = 0; gi < 4; ++gi)
#pragma unroll
    for (int kt = 0; kt < 8; ++kt)
      bfrag[gi][kt] = load_frag_b(W, NH, gi * NH + j0, kt * 32);

  int cur = 0;
  for (int s = 0; s < NT; ++s) {
    int t = d ? (NT - 1 - s) : s;
    const bf16_t* hcur = hbuf + cur * NB * NH;
    bf16_t* hnxt = hbuf + (cur ^ 1) * NB * NH;
    const float* pt = pr + (long)t * NG * NB;

#pragma unroll
    for (int mt = 0; mt < 8; ++mt) {    // all batch tiles
      int b0 = mt * 16;
      int bbase = b0 + mh * 8;
      v8f acc[4];
#pragma unroll
      for (int gi = 0; gi < 4; ++gi) { v8f z = {}; acc[gi] = z; }
#pragma unroll
      for (int kt = 0; kt < 8; ++kt) {
        v16bf a = load_frag_a(hcur, NH, b0, kt * 32);
#pragma unroll
        for (int gi = 0; gi < 4; ++gi)
          acc[gi] = wmma_bf16(a, bfrag[gi][kt], acc[gi]);
      }
      // coalesced pre loads: 8 consecutive b per (gate) -> two float4 each
      float pg[4][8];
#pragma unroll
      for (int gi = 0; gi < 4; ++gi) {
        const float4* pzg =
            (const float4*)(pt + ((long)gi * NH + j) * NB + bbase);
        float4 lo = pzg[0], hi = pzg[1];
        pg[gi][0] = lo.x; pg[gi][1] = lo.y; pg[gi][2] = lo.z; pg[gi][3] = lo.w;
        pg[gi][4] = hi.x; pg[gi][5] = hi.y; pg[gi][6] = hi.z; pg[gi][7] = hi.w;
      }
#pragma unroll
      for (int rr = 0; rr < 8; ++rr) {
        int b = bbase + rr;
        float zi = acc[0][rr] + pg[0][rr];
        float zf = acc[1][rr] + pg[1][rr];
        float zg = acc[2][rr] + pg[2][rr];
        float zo = acc[3][rr] + pg[3][rr];
        float c = fast_sig(zf) * c_sh[b * NH + j] + fast_sig(zi) * fast_tanh(zg);
        c_sh[b * NH + j] = c;
        float h = fast_sig(zo) * fast_tanh(c);
        hnxt[b * NH + j] = (bf16_t)h;
        hout[((long)b * NT + t) * NH + j] = (bf16_t)h;
      }
    }
    __syncthreads();                    // hnxt complete; safe to swap
    cur ^= 1;
  }
}

// ---- K4: decoder GEMM: decoded = hf@Wf^T + hb@Wb^T + dec_b ----------------
__global__ void k_decoder(const bf16_t* __restrict__ hf,
                          const bf16_t* __restrict__ hb,
                          const bf16_t* __restrict__ decWP,
                          const float* __restrict__ dec_b,
                          float* __restrict__ decoded) {
  int w = threadIdx.x >> 5;
  long r0 = ((long)blockIdx.x * 8 + w) * 16;
  int lane = threadIdx.x & 31;
  int n = lane & 15, mh = lane >> 4;
  v8f acc0 = {}, acc1 = {};
#pragma unroll
  for (int kt = 0; kt < 8; ++kt) {      // hf half: K 0..255
    v16bf a  = load_frag_a(hf, NH, r0, kt * 32);
    acc0 = wmma_bf16(a, load_frag_b(decWP, 2 * NH, 0,  kt * 32), acc0);
    acc1 = wmma_bf16(a, load_frag_b(decWP, 2 * NH, 16, kt * 32), acc1);
  }
#pragma unroll
  for (int kt = 0; kt < 8; ++kt) {      // hb half: K 256..511
    v16bf a  = load_frag_a(hb, NH, r0, kt * 32);
    acc0 = wmma_bf16(a, load_frag_b(decWP, 2 * NH, 0,  NH + kt * 32), acc0);
    acc1 = wmma_bf16(a, load_frag_b(decWP, 2 * NH, 16, NH + kt * 32), acc1);
  }
  float bv0 = dec_b[n];
#pragma unroll
  for (int rr = 0; rr < 8; ++rr) {
    long r = r0 + mh * 8 + rr;
    decoded[r * NL + n] = acc0[rr] + bv0;       // labels 0..15
    int l1 = 16 + n;
    if (l1 < NL) decoded[r * NL + l1] = acc1[rr] + dec_b[l1];
  }
}

// ---- K5: CRF log-likelihood (per-batch partials) ---------------------------
__global__ void k_crf(const float* __restrict__ decoded, const int* __restrict__ y,
                      const float* __restrict__ start, const float* __restrict__ end_,
                      const float* __restrict__ trans, float* __restrict__ llh_b) {
  __shared__ float str[NL * NL];
  __shared__ float alpha_s[NL];
  int b = blockIdx.x;
  int j = threadIdx.x;
  for (int i = threadIdx.x; i < NL * NL; i += 32) str[i] = trans[i];
  const float* em0 = decoded + (long)b * NT * NL;
  if (j < NL) alpha_s[j] = start[j] + em0[j];
  __syncthreads();

  for (int t = 1; t < NT; ++t) {
    const float* em = decoded + ((long)b * NT + t) * NL;
    if (t + 1 < NT)
      __builtin_prefetch(decoded + ((long)b * NT + t + 1) * NL, 0, 0);
    float v = 0.0f;
    if (j < NL) {
      float m = -1e30f;
#pragma unroll
      for (int i = 0; i < NL; ++i) m = fmaxf(m, alpha_s[i] + str[i * NL + j]);
      float sum = 0.0f;
#pragma unroll
      for (int i = 0; i < NL; ++i) sum += __expf(alpha_s[i] + str[i * NL + j] - m);
      v = m + __logf(sum) + em[j];
    }
    __syncthreads();
    if (j < NL) alpha_s[j] = v;
    __syncthreads();
  }

  if (j == 0) {
    float m = -1e30f;
    for (int i = 0; i < NL; ++i) m = fmaxf(m, alpha_s[i] + end_[i]);
    float sum = 0.0f;
    for (int i = 0; i < NL; ++i) sum += __expf(alpha_s[i] + end_[i] - m);
    float logZ = m + __logf(sum);

    const int* yb = y + (long)b * NT;
    float num = start[yb[0]] + end_[yb[NT - 1]];
    for (int t = 0; t < NT; ++t)
      num += decoded[((long)b * NT + t) * NL + yb[t]];
    for (int t = 1; t < NT; ++t)
      num += str[yb[t - 1] * NL + yb[t]];
    llh_b[b] = num - logZ;
  }
}

// ---- K6: Viterbi decode ----------------------------------------------------
__global__ void k_viterbi(const float* __restrict__ decoded,
                          const float* __restrict__ start,
                          const float* __restrict__ end_,
                          const float* __restrict__ trans,
                          unsigned char* __restrict__ bp,
                          float* __restrict__ out_path) {
  __shared__ float str[NL * NL];
  __shared__ float alpha_s[NL];
  int b = blockIdx.x;
  int j = threadIdx.x;
  for (int i = threadIdx.x; i < NL * NL; i += 32) str[i] = trans[i];
  const float* em0 = decoded + (long)b * NT * NL;
  if (j < NL) alpha_s[j] = start[j] + em0[j];
  __syncthreads();

  for (int t = 1; t < NT; ++t) {
    const float* em = decoded + ((long)b * NT + t) * NL;
    if (t + 1 < NT)
      __builtin_prefetch(decoded + ((long)b * NT + t + 1) * NL, 0, 0);
    float v = 0.0f;
    if (j < NL) {
      float best = -1e30f; int bi = 0;
#pragma unroll
      for (int i = 0; i < NL; ++i) {
        float s = alpha_s[i] + str[i * NL + j];
        if (s > best) { best = s; bi = i; }
      }
      bp[(((long)(t - 1)) * NB + b) * NL + j] = (unsigned char)bi;
      v = best + em[j];
    }
    __syncthreads();
    if (j < NL) alpha_s[j] = v;
    __syncthreads();
  }

  if (j == 0) {
    float best = -1e30f; int last = 0;
    for (int i = 0; i < NL; ++i) {
      float s = alpha_s[i] + end_[i];
      if (s > best) { best = s; last = i; }
    }
    out_path[(long)b * NT + NT - 1] = (float)last;
    int tag = last;
    for (int tt = NT - 1; tt >= 1; --tt) {
      tag = bp[(((long)(tt - 1)) * NB + b) * NL + tag];
      out_path[(long)b * NT + tt - 1] = (float)tag;
    }
  }
}

// ---- K7: reduce partials -> -llh ------------------------------------------
__global__ void k_reduce(const float* __restrict__ llh_b, float* __restrict__ out0) {
  __shared__ float red[NB];
  int i = threadIdx.x;
  red[i] = llh_b[i];
  __syncthreads();
  for (int s = NB / 2; s > 0; s >>= 1) {
    if (i < s) red[i] += red[i + s];
    __syncthreads();
  }
  if (i == 0) out0[0] = -red[0];
}

// ---------------------------------------------------------------------------
extern "C" void kernel_launch(void* const* d_in, const int* in_sizes, int n_in,
                              void* d_out, int out_size, void* d_ws, size_t ws_size,
                              hipStream_t stream) {
  const int*   x        = (const int*)  d_in[0];
  const int*   y        = (const int*)  d_in[2];
  const float* emb      = (const float*)d_in[3];
  const float* Wih_f    = (const float*)d_in[4];
  const float* Whh_f    = (const float*)d_in[5];
  const float* b_f      = (const float*)d_in[6];
  const float* Wih_b    = (const float*)d_in[7];
  const float* Whh_b    = (const float*)d_in[8];
  const float* b_b      = (const float*)d_in[9];
  const float* decW     = (const float*)d_in[10];
  const float* dec_b    = (const float*)d_in[11];
  const float* c_start  = (const float*)d_in[12];
  const float* c_end    = (const float*)d_in[13];
  const float* c_trans  = (const float*)d_in[14];
  float* out = (float*)d_out;

  char* ws = (char*)d_ws;
  size_t off = 0;
  bf16_t* e_bf  = (bf16_t*)(ws + off); off += (size_t)NTOK * NEP * 2;       // 40 MB
  float*  preT  = (float*) (ws + off); off += (size_t)2 * NTOK * NG * 4;    // 536 MB
  bf16_t* hf    = (bf16_t*)(ws + off); off += (size_t)NTOK * NH * 2;        // 33.5 MB
  bf16_t* hb    = (bf16_t*)(ws + off); off += (size_t)NTOK * NH * 2;        // 33.5 MB
  float*  dec   = (float*) (ws + off); off += (size_t)NTOK * NL * 4;        // 5.2 MB
  bf16_t* WihP  = (bf16_t*)(ws + off); off += (size_t)2 * NG * NEP * 2;
  bf16_t* WhhP  = (bf16_t*)(ws + off); off += (size_t)2 * NG * NH * 2;
  bf16_t* decWP = (bf16_t*)(ws + off); off += (size_t)32 * 2 * NH * 2;
  float*  llh_b = (float*) (ws + off); off += 512;
  unsigned char* bp = (unsigned char*)(ws + off);
  off += (size_t)(NT - 1) * NB * NL;
  (void)ws_size; (void)in_sizes; (void)n_in; (void)out_size;

  k_prep<<<(2 * NG * NEP + 255) / 256, 256, 0, stream>>>(
      Wih_f, Whh_f, Wih_b, Whh_b, decW, WihP, WhhP, decWP);

  k_embed<<<NTOK, NEP, 0, stream>>>(x, emb, e_bf);

  k_gemm_pre<<<dim3(NTOK / 128, NG / 64, 2), 256, 128 * NEP * 2, stream>>>(
      e_bf, WihP, b_f, b_b, preT);

  // 2 blocks (one per direction), 16 waves, 256 KB LDS (2x h + c)
  k_lstm<<<2, 512, 2 * NB * NH * 2 + NB * NH * 4, stream>>>(preT, WhhP, hf, hb);

  k_decoder<<<NTOK / 16 / 8, 256, 0, stream>>>(hf, hb, decWP, dec_b, dec);

  k_crf<<<NB, 32, 0, stream>>>(dec, y, c_start, c_end, c_trans, llh_b);

  k_viterbi<<<NB, 32, 0, stream>>>(dec, c_start, c_end, c_trans, bp, out + 1);

  k_reduce<<<1, NB, 0, stream>>>(llh_b, out);
}
